// Reacher_DeLaN_Network_18854906429608
// MI455X (gfx1250) — matrix operand order repaired and checked by reading, not compiled, not run on hardware
//
#include <hip/hip_runtime.h>
#include <hip/hip_bf16.h>

typedef __attribute__((ext_vector_type(16))) _Float16 v16h;
typedef __attribute__((ext_vector_type(8)))  float    v8f;

#define HDIM 64
#define TILE 16

union FragU { uint4 u[2]; v16h h; };

// A-matrix fragment (16x32 f16, documented layout):
// lanes 0-15: K = c*32 + {0..7, 16..23}; lanes 16-31: +8
__device__ __forceinline__ v16h ldsFragA(const _Float16* row, int c, int half) {
  FragU f;
  const _Float16* p = row + c * 32 + half * 8;
  f.u[0] = *(const uint4*)(p);
  f.u[1] = *(const uint4*)(p + 16);
  return f.h;
}
// B-matrix fragment (32x16 f16): lanes 0-15 hold K = c*32 + {0..7, 16..23},
// lanes 16-31: +8 (same order as the A fragment)
__device__ __forceinline__ v16h ldsFragB(const _Float16* row, int c, int half) {
  FragU f;
  const _Float16* p = row + c * 32 + half * 8;
  f.u[0] = *(const uint4*)(p);
  f.u[1] = *(const uint4*)(p + 16);
  return f.h;
}
__device__ __forceinline__ unsigned pack2(float a, float b) {
  union { _Float16 h[2]; unsigned u; } p;
  p.h[0] = (_Float16)a; p.h[1] = (_Float16)b; return p.u;
}
__device__ __forceinline__ v8f wmma_f16(v16h a, v16h b, v8f c) {
  return __builtin_amdgcn_wmma_f32_16x16x32_f16(false, a, false, b, (short)0, c, false, false);
}

__global__ __launch_bounds__(256) void delan_reacher_kernel(
    const float* __restrict__ x,
    const float* __restrict__ W1,  const float* __restrict__ b1,
    const float* __restrict__ W1a, const float* __restrict__ b1a,
    const float* __restrict__ W2,  const float* __restrict__ b2,
    const float* __restrict__ W3,  const float* __restrict__ b3,
    const float* __restrict__ W4,  const float* __restrict__ b4,
    float* __restrict__ out, int N)
{
  __shared__ _Float16 sW1a[HDIM * HDIM];      // row-major W1a[n][k], f16
  __shared__ _Float16 sWp[16 * HDIM];         // rows: W2(2), W3(2), W4(1), zeros
  __shared__ float sW1[HDIM * 2];
  __shared__ float sB1[HDIM];
  __shared__ float sB1a[HDIM];
  __shared__ float sBsm[8];                   // b2(2), b3(2), b4(1)
  __shared__ __align__(16) _Float16 sStage[8][3 * TILE * HDIM]; // per-wave staging

  const int t = threadIdx.x;
  for (int i = t; i < HDIM * HDIM; i += 256) sW1a[i] = (_Float16)W1a[i];
  for (int i = t; i < 16 * HDIM; i += 256) {
    int r = i >> 6, k = i & 63;
    float v = 0.f;
    if (r < 2)      v = W2[r * HDIM + k];
    else if (r < 4) v = W3[(r - 2) * HDIM + k];
    else if (r == 4) v = W4[k];
    sWp[i] = (_Float16)v;
  }
  if (t < 128) sW1[t] = W1[t];
  if (t < 64) { sB1[t] = b1[t]; sB1a[t] = b1a[t]; }
  if (t < 2) { sBsm[t] = b2[t]; sBsm[2 + t] = b3[t]; }
  if (t == 0) { sBsm[4] = b4[0]; sBsm[5] = 0.f; sBsm[6] = 0.f; sBsm[7] = 0.f; }
  __syncthreads();

  const int lane = t & 31;
  const int wave = t >> 5;
  const int half = lane >> 4;
  const int lm   = lane & 15;

  // Preload B fragments (reused every tile)
  v16h Bw[2][4];
  v16h Bp[2];
  #pragma unroll
  for (int c = 0; c < 2; ++c) {
    #pragma unroll
    for (int nt = 0; nt < 4; ++nt)
      Bw[c][nt] = ldsFragB(&sW1a[(nt * 16 + lm) * HDIM], c, half);
    Bp[c] = ldsFragB(&sWp[lm * HDIM], c, half);
  }

  _Float16* stg   = sStage[wave];
  _Float16* stgH  = stg;
  _Float16* stgV0 = stg + TILE * HDIM;
  _Float16* stgV1 = stg + 2 * TILE * HDIM;

  const int nTiles = N >> 4;
  const int waveId = blockIdx.x * 8 + wave;
  const int nWaves = gridDim.x * 8;
  float* outTau = out;
  float* outHm  = out + (size_t)2 * N;
  float* outC   = out + (size_t)6 * N;
  float* outG   = out + (size_t)8 * N;
  const v8f vzero = {};

  for (int tile = waveId; tile < nTiles; tile += nWaves) {
    const int s = tile * 16 + lm;
    const float* xs = x + (size_t)s * 6;
    float2 xq   = *(const float2*)(xs);
    float2 xqd  = *(const float2*)(xs + 2);
    float2 xqdd = *(const float2*)(xs + 4);
    const float q0 = xq.x,   q1 = xq.y;
    const float qd0 = xqd.x, qd1 = xqd.y;
    const float qdd0 = xqdd.x, qdd1 = xqdd.y;

    // ---- layer 1 (VALU): h1, v0 = dR1*W1[:,0], v1 = dR1*W1[:,1] -> f16 staging
    {
      unsigned* dH  = (unsigned*)&stgH [lm * HDIM + half * 32];
      unsigned* dV0 = (unsigned*)&stgV0[lm * HDIM + half * 32];
      unsigned* dV1 = (unsigned*)&stgV1[lm * HDIM + half * 32];
      #pragma unroll
      for (int jj = 0; jj < 16; ++jj) {
        const int h0 = half * 32 + 2 * jj;
        const float w00 = sW1[h0 * 2 + 0], w01 = sW1[h0 * 2 + 1];
        const float w10 = sW1[h0 * 2 + 2], w11 = sW1[h0 * 2 + 3];
        const float z0 = fmaf(w00, q0, fmaf(w01, q1, sB1[h0]));
        const float z1 = fmaf(w10, q0, fmaf(w11, q1, sB1[h0 + 1]));
        const float a0 = z0 > 0.f ? z0 : 0.01f * z0;
        const float a1 = z1 > 0.f ? z1 : 0.01f * z1;
        const float d0 = z0 > 0.f ? 1.f : -0.01f;   // reference uses NEG_SLOPE=-0.01 in the mask
        const float d1 = z1 > 0.f ? 1.f : -0.01f;
        dH [jj] = pack2(a0, a1);
        dV0[jj] = pack2(d0 * w00, d1 * w10);
        dV1[jj] = pack2(d0 * w01, d1 * w11);
      }
    }
    asm volatile("s_wait_dscnt 0" ::: "memory");

    // ---- level-1 GEMMs: [h1; v0; v1](16x64) @ W1a^T(64x64) via 24 WMMAs
    v8f accH[4], accD0[4], accD1[4];
    #pragma unroll
    for (int nt = 0; nt < 4; ++nt) { accH[nt] = vzero; accD0[nt] = vzero; accD1[nt] = vzero; }
    #pragma unroll
    for (int c = 0; c < 2; ++c) {
      v16h Ah = ldsFragA(&stgH [lm * HDIM], c, half);
      v16h A0 = ldsFragA(&stgV0[lm * HDIM], c, half);
      v16h A1 = ldsFragA(&stgV1[lm * HDIM], c, half);
      #pragma unroll
      for (int nt = 0; nt < 4; ++nt) {
        accH [nt] = wmma_f16(Ah, Bw[c][nt], accH [nt]);
        accD0[nt] = wmma_f16(A0, Bw[c][nt], accD0[nt]);
        accD1[nt] = wmma_f16(A1, Bw[c][nt], accD1[nt]);
      }
    }

    // ---- bias + leaky + dR2 mask; restage h2, dh2_dq as f16 (overwrite staging)
    #pragma unroll
    for (int nt = 0; nt < 4; ++nt) {
      const int n = nt * 16 + lm;
      const float bn = sB1a[n];
      #pragma unroll
      for (int r = 0; r < 8; ++r) {
        const int m = half * 8 + r;           // C/D layout: lanes 16-31 are rows +8
        const float z = accH[nt][r] + bn;
        const float h2 = z > 0.f ? z : 0.01f * z;
        const float dr = z > 0.f ? 1.f : -0.01f;
        stgH [m * HDIM + n] = (_Float16)h2;
        stgV0[m * HDIM + n] = (_Float16)(dr * accD0[nt][r]);
        stgV1[m * HDIM + n] = (_Float16)(dr * accD1[nt][r]);
      }
    }
    asm volatile("s_wait_dscnt 0" ::: "memory");

    // ---- level-2: project onto [W2;W3;W4] (5 useful cols) — 6 WMMAs
    v8f pH = vzero, p0 = vzero, p1 = vzero;
    #pragma unroll
    for (int c = 0; c < 2; ++c) {
      pH = wmma_f16(ldsFragA(&stgH [lm * HDIM], c, half), Bp[c], pH);
      p0 = wmma_f16(ldsFragA(&stgV0[lm * HDIM], c, half), Bp[c], p0);
      p1 = wmma_f16(ldsFragA(&stgV1[lm * HDIM], c, half), Bp[c], p1);
    }

    float* P = (float*)stg;                   // reuse staging: 3 x 16x16 f32
    #pragma unroll
    for (int r = 0; r < 8; ++r) {
      const int m = half * 8 + r;
      P[0 * 256 + m * 16 + lm] = pH[r];
      P[1 * 256 + m * 16 + lm] = p0[r];
      P[2 * 256 + m * 16 + lm] = p1[r];
    }
    asm volatile("s_wait_dscnt 0" ::: "memory");

    // ---- closed-form 2x2 DeLaN algebra, one sample per lane (lm)
    const float g0   = P[0 * 256 + lm * 16 + 0] + sBsm[0];
    const float g1   = P[0 * 256 + lm * 16 + 1] + sBsm[1];
    const float ldp0 = P[0 * 256 + lm * 16 + 2] + sBsm[2];
    const float ldp1 = P[0 * 256 + lm * 16 + 3] + sBsm[3];
    const float lo   = P[0 * 256 + lm * 16 + 4] + sBsm[4];
    const float w30d0 = P[1 * 256 + lm * 16 + 2];
    const float w31d0 = P[1 * 256 + lm * 16 + 3];
    const float w4d0  = P[1 * 256 + lm * 16 + 4];
    const float w30d1 = P[2 * 256 + lm * 16 + 2];
    const float w31d1 = P[2 * 256 + lm * 16 + 3];
    const float w4d1  = P[2 * 256 + lm * 16 + 4];

    const float ld0 = fmaxf(ldp0, 0.f), ld1 = fmaxf(ldp1, 0.f);
    const float m0 = ldp0 > 0.f ? 1.f : 0.f;
    const float m1 = ldp1 > 0.f ? 1.f : 0.f;
    const float dld00 = m0 * w30d0, dld01 = m0 * w30d1;
    const float dld10 = m1 * w31d0, dld11 = m1 * w31d1;
    const float dlo0 = w4d0, dlo1 = w4d1;

    const float a0  = dld00 * qd0 + dld01 * qd1;   // dld_dt[0]
    const float a1t = dld10 * qd0 + dld11 * qd1;   // dld_dt[1]
    const float b0  = dlo0 * qd0 + dlo1 * qd1;     // dlo_dt[0]

    const float H00 = ld0 * ld0 + 1e-5f;
    const float H01 = ld0 * lo;
    const float H11 = lo * lo + ld1 * ld1 + 1e-5f;

    const float dH00 = 2.f * ld0 * a0;
    const float dH01 = ld0 * b0 + a0 * lo;
    const float dH11 = 2.f * (lo * b0 + ld1 * a1t);

    const float qq00 = qd0 * qd0, qq01 = qd0 * qd1, qq11 = qd1 * qd1;
    const float quad0 = 2.f * (dld00 * ld0 * qq00 + (dld00 * lo + dlo0 * ld0) * qq01
                               + (dlo0 * lo + dld10 * ld1) * qq11);
    const float quad1 = 2.f * (dld01 * ld0 * qq00 + (dld01 * lo + dlo1 * ld0) * qq01
                               + (dlo1 * lo + dld11 * ld1) * qq11);

    const float c0 = dH00 * qd0 + dH01 * qd1 - 0.5f * quad0;
    const float c1 = dH01 * qd0 + dH11 * qd1 - 0.5f * quad1;
    const float tau0 = H00 * qdd0 + H01 * qdd1 + c0 + g0;
    const float tau1 = H01 * qdd0 + H11 * qdd1 + c1 + g1;

    if (half == 0) {
      outTau[(size_t)2 * s + 0] = tau0;
      outTau[(size_t)2 * s + 1] = tau1;
      outHm [(size_t)4 * s + 0] = H00;
      outHm [(size_t)4 * s + 1] = H01;
      outHm [(size_t)4 * s + 2] = H01;
      outHm [(size_t)4 * s + 3] = H11;
      outC  [(size_t)2 * s + 0] = c0;
      outC  [(size_t)2 * s + 1] = c1;
      outG  [(size_t)2 * s + 0] = g0;
      outG  [(size_t)2 * s + 1] = g1;
    }
  }
}

extern "C" void kernel_launch(void* const* d_in, const int* in_sizes, int n_in,
                              void* d_out, int out_size, void* d_ws, size_t ws_size,
                              hipStream_t stream) {
  const float* x   = (const float*)d_in[0];
  const float* W1  = (const float*)d_in[1];
  const float* b1  = (const float*)d_in[2];
  const float* W1a = (const float*)d_in[3];
  const float* b1a = (const float*)d_in[4];
  const float* W2  = (const float*)d_in[5];
  const float* b2  = (const float*)d_in[6];
  const float* W3  = (const float*)d_in[7];
  const float* b3  = (const float*)d_in[8];
  const float* W4  = (const float*)d_in[9];
  const float* b4  = (const float*)d_in[10];
  float* out = (float*)d_out;

  const int N = in_sizes[0] / 6;
  const int nTiles = N >> 4;                 // 16 samples per wave-tile
  int blocks = (nTiles + 7) / 8;             // 8 waves per block, 1 tile per wave
  if (blocks > 2048) blocks = 2048;
  if (blocks < 1) blocks = 1;

  delan_reacher_kernel<<<blocks, 256, 0, stream>>>(
      x, W1, b1, W1a, b1a, W2, b2, W3, b3, W4, b4, out, N);
}
